// DFMNLoss_38551626449439
// MI455X (gfx1250) — matrix operand order repaired and checked by reading, not compiled, not run on hardware
//
#include <hip/hip_runtime.h>
#include <hip/hip_bf16.h>

// ---- problem constants (match reference) ----
#define BATCH 512
#define CDIM  1024
#define FSQ   196      // 14*14 pixels per image
#define NWAY  64
#define NPIXTILES 13   // ceil(196/16)

typedef __attribute__((ext_vector_type(16))) __bf16 v16bf;
typedef __attribute__((ext_vector_type(8)))  float  v8f;
typedef __attribute__((ext_vector_type(4)))  unsigned int v4u;

// -----------------------------------------------------------------------------
// Kernel 1: gather prototypes via `indices`, convert to bf16, compute ||p||^2.
// 64 blocks, one episode class each.
// -----------------------------------------------------------------------------
__global__ __launch_bounds__(256) void protos_prep(const float* __restrict__ prototypes,
                                                   const int* __restrict__ indices,
                                                   __bf16* __restrict__ pbf,
                                                   float* __restrict__ pn) {
    int i = blockIdx.x;                 // episode class 0..63
    int row = indices[i];               // global prototype row
    const float* src = prototypes + (size_t)row * CDIM;
    float part = 0.f;
    for (int k = threadIdx.x; k < CDIM; k += 256) {
        float v = src[k];
        pbf[(size_t)i * CDIM + k] = (__bf16)v;
        part += v * v;
    }
    __shared__ float red[256];
    red[threadIdx.x] = part;
    __syncthreads();
    for (int s = 128; s > 0; s >>= 1) {
        if (threadIdx.x < s) red[threadIdx.x] += red[threadIdx.x + s];
        __syncthreads();
    }
    if (threadIdx.x == 0) pn[i] = red[0];
}

// -----------------------------------------------------------------------------
// Kernel 2: per-image WMMA GEMM + fused log-softmax loss.
// Grid = 512 blocks (one image), 256 threads = 8 waves (wave32).
// Each wave owns pixel tiles (16 pixels) and all 4 proto tiles (4x v8f accum).
// D = A(protos 16x32 bf16) x B(q^T 32x16 bf16), accumulate f32 over K=1024.
// -----------------------------------------------------------------------------
__global__ __launch_bounds__(256) void dfmn_main(const float* __restrict__ qf,
                                                 const int* __restrict__ labels,
                                                 const __bf16* __restrict__ pbf,
                                                 const float* __restrict__ pn_g,
                                                 float* __restrict__ partials) {
    const int b    = blockIdx.x;
    const int tid  = threadIdx.x;
    const int lane = tid & 31;
    const int w    = tid >> 5;
    const int nloc = lane & 15;             // column (pixel-in-tile / proto-row lane)
    const int hi   = (lane >= 16) ? 1 : 0;  // lane half

    __shared__ float pn_s[NWAY];
    __shared__ float wsum[8];
    if (tid < NWAY) pn_s[tid] = pn_g[tid];
    __syncthreads();

    const int label = labels[b];
    const size_t imgBase = (size_t)b * (size_t)(CDIM * FSQ);

    float lane_acc = 0.f;

    for (int tile = w; tile < NPIXTILES; tile += 8) {
        const int pixel  = tile * 16 + nloc;
        const bool valid = (pixel < FSQ);
        const int pclamp = valid ? pixel : (FSQ - 1);
        const float* qb  = qf + imgBase + pclamp;   // q[pixel][k] at stride FSQ

        v8f acc[4] = {};   // 4 proto tiles x (16x16 f32 / 8 VGPRs)

        for (int k0 = 0; k0 < CDIM; k0 += 32) {
            // ---- B fragment: q^T 32x16 bf16. lane half selects K base +0 / +8;
            //      elements 0-7: K=kbB..kbB+7, elements 8-15: K=kbB+16..kbB+23.
            const int kbB = k0 + (hi ? 8 : 0);
            v16bf bf;
            #pragma unroll
            for (int j = 0; j < 16; ++j)
                bf[j] = (__bf16)qb[(size_t)(kbB + j + (j >= 8 ? 8 : 0)) * FSQ];   // coalesced per-k dwords

            // ---- A fragments: protos 16x32 bf16 (documented layout):
            //      lane half selects K base +0 / +8; VGPR0-3: K=base..base+7,
            //      VGPR4-7: K=base+16..base+23. Two 16B loads per fragment.
            const int kbA = k0 + (hi ? 8 : 0);
            #pragma unroll
            for (int t = 0; t < 4; ++t) {
                const __bf16* pa = pbf + (size_t)(t * 16 + nloc) * CDIM + kbA;
                union { v16bf v; v4u u[2]; } a;
                a.u[0] = *(const v4u*)(pa);        // K = kbA .. kbA+7
                a.u[1] = *(const v4u*)(pa + 16);   // K = kbA+16 .. kbA+23
                acc[t] = __builtin_amdgcn_wmma_f32_16x16x32_bf16(
                    false, a.v, false, bf, (short)0, acc[t], false, false);
            }
        }

        // ---- s[m] = 2*dot - ||p_m||^2 ; lane pair (L, L+16) holds all 64 m.
        float sv[4][8];
        float mx = -3.402823466e38f;
        #pragma unroll
        for (int t = 0; t < 4; ++t)
            #pragma unroll
            for (int v = 0; v < 8; ++v) {
                const int m = t * 16 + v + (hi ? 8 : 0);
                float s = 2.f * acc[t][v] - pn_s[m];
                sv[t][v] = s;
                mx = fmaxf(mx, s);
            }
        mx = fmaxf(mx, __shfl_xor(mx, 16, 32));

        float sum = 0.f, slab = 0.f;
        #pragma unroll
        for (int t = 0; t < 4; ++t)
            #pragma unroll
            for (int v = 0; v < 8; ++v) {
                const int m = t * 16 + v + (hi ? 8 : 0);
                sum += __expf(sv[t][v] - mx);
                if (m == label) slab = sv[t][v];
            }
        sum  += __shfl_xor(sum, 16, 32);
        slab += __shfl_xor(slab, 16, 32);

        const float lse = mx + __logf(sum);
        if (valid && !hi) lane_acc += (lse - slab);   // count each pixel once
    }

    // wave reduction, then block reduction -> per-image partial
    #pragma unroll
    for (int o = 16; o >= 1; o >>= 1) lane_acc += __shfl_xor(lane_acc, o, 32);
    if (lane == 0) wsum[w] = lane_acc;
    __syncthreads();
    if (tid == 0) {
        float tot = 0.f;
        #pragma unroll
        for (int i = 0; i < 8; ++i) tot += wsum[i];
        partials[b] = tot;
    }
}

// -----------------------------------------------------------------------------
// Kernel 3: deterministic final reduction of 512 per-image partials.
// -----------------------------------------------------------------------------
__global__ __launch_bounds__(512) void finalize(const float* __restrict__ partials,
                                                float* __restrict__ out) {
    __shared__ float red[512];
    red[threadIdx.x] = partials[threadIdx.x];
    __syncthreads();
    for (int s = 256; s > 0; s >>= 1) {
        if (threadIdx.x < s) red[threadIdx.x] += red[threadIdx.x + s];
        __syncthreads();
    }
    if (threadIdx.x == 0) out[0] = red[0] / (float)(BATCH * FSQ);
}

// -----------------------------------------------------------------------------
extern "C" void kernel_launch(void* const* d_in, const int* in_sizes, int n_in,
                              void* d_out, int out_size, void* d_ws, size_t ws_size,
                              hipStream_t stream) {
    const float* qf         = (const float*)d_in[0];   // [512,1024,14,14]
    const int*   labels     = (const int*)d_in[1];     // [512]
    const float* prototypes = (const float*)d_in[2];   // [100,1024]
    const int*   indices    = (const int*)d_in[3];     // [64]
    float*       out        = (float*)d_out;           // scalar loss

    // scratch layout: [pn: 64 f32 | pad to 256B | protos bf16: 64*1024 | partials: 512 f32]
    float*  ws_pn   = (float*)d_ws;
    __bf16* ws_pbf  = (__bf16*)((char*)d_ws + 256);
    float*  ws_part = (float*)((char*)d_ws + 256 + (size_t)NWAY * CDIM * sizeof(__bf16));

    protos_prep<<<NWAY, 256, 0, stream>>>(prototypes, indices, ws_pbf, ws_pn);
    dfmn_main<<<BATCH, 256, 0, stream>>>(qf, labels, ws_pbf, ws_pn, ws_part);
    finalize<<<1, 512, 0, stream>>>(ws_part, out);
}
